// Attention_32469952758511
// MI455X (gfx1250) — matrix operand order repaired and checked by reading, not compiled, not run on hardware
//
#include <hip/hip_runtime.h>

// ---------------------------------------------------------------------------
// MLA forward for MI455X (gfx1250): bf16 WMMA (v_wmma_f32_16x16x32_bf16) with
// f32 accumulation, async global->LDS double-buffered weight staging in the
// GEMMs, flash-style causal attention (4 waves / block, one 16-query tile per
// wave).
// ---------------------------------------------------------------------------

typedef float  v8f   __attribute__((ext_vector_type(8)));
typedef __bf16 v8bf  __attribute__((ext_vector_type(8)));
typedef __bf16 v16bf __attribute__((ext_vector_type(16)));
typedef int    v4i   __attribute__((vector_size(16)));

constexpr int Bc    = 4;
constexpr int Sc    = 1024;
constexpr int Hc    = 16;
constexpr int HIDc  = 2048;
constexpr int QLRc  = 1536;
constexpr int KVLRc = 512;
constexpr int NOPEc = 128;
constexpr int ROPEc = 64;
constexpr int VDc   = 128;
constexpr int DQKc  = NOPEc + ROPEc;   // 192
constexpr int BSc   = Bc * Sc;         // 4096

union BF16x16 { v16bf v; v8bf h[2]; };

// A-operand fragment (16x32 bf16): lane holds row = lane%16,
// K = kb..kb+15 contiguous, kb = kk + 16*(lane>>4).  Two b128 loads.
__device__ __forceinline__ v16bf load_afrag(const __bf16* row, int kb) {
  BF16x16 t;
  t.h[0] = *(const v8bf*)(row + kb);
  t.h[1] = *(const v8bf*)(row + kb + 8);
  return t.v;
}

// B-operand fragment (32x16 bf16): lane holds col = lane%16,
// K = kk + 16*(lane>>4) .. +15 contiguous.  One 32-byte load.
__device__ __forceinline__ v16bf load_bfrag(const __bf16* row, int kb) {
  return *(const v16bf*)(row + kb);
}

__device__ __forceinline__ v8f wmma_bf16(v16bf a, v16bf b, v8f c) {
  return __builtin_amdgcn_wmma_f32_16x16x32_bf16(false, a, false, b, (short)0, c,
                                                 false, false);
}

// ---------------------------------------------------------------------------
// Async global -> LDS staging (gfx1250 GLOBAL_LOAD_ASYNC_TO_LDS_B128), with a
// compile-safe fallback (load + ds_store) if the builtin is absent.
// The builtin takes pointers to 16-byte int vectors in AS1 (global) / AS3 (LDS).
// ---------------------------------------------------------------------------
#if __has_builtin(__builtin_amdgcn_global_load_async_to_lds_b128) && \
    __has_builtin(__builtin_amdgcn_s_wait_asynccnt)
#define HAVE_ASYNC_LDS 1
#else
#define HAVE_ASYNC_LDS 0
#endif

__device__ __forceinline__ void stage16(const __bf16* g, __bf16* l) {
#if HAVE_ASYNC_LDS
  __builtin_amdgcn_global_load_async_to_lds_b128(
      (__attribute__((address_space(1))) v4i*)(v4i*)g,
      (__attribute__((address_space(3))) v4i*)(v4i*)l, 0, 0);
#else
  *(uint4*)l = *(const uint4*)g;
#endif
}

__device__ __forceinline__ void stage_wait() {
#if HAVE_ASYNC_LDS
  __builtin_amdgcn_s_wait_asynccnt(0);
#endif
}

// ---------------------------------------------------------------------------
// f32 -> bf16 conversion
// ---------------------------------------------------------------------------
__global__ __launch_bounds__(256)
void cvt_bf16(const float* __restrict__ in, __bf16* __restrict__ out, int n) {
  int i = blockIdx.x * 256 + threadIdx.x;
  if (i < n) out[i] = (__bf16)in[i];
}

// ---------------------------------------------------------------------------
// C[m,n] = sum_k A[m,k] * W[n,k]   (A: MxK bf16 rm, W: NxK bf16 rm)
// 4 waves / block; each wave: 16 rows x 64 cols; all waves share one 64-col
// W slab staged per 32-k step into double-buffered LDS via async loads.
// grid = (N/64, M/64).
// ---------------------------------------------------------------------------
template <bool OUT_BF16>
__global__ __launch_bounds__(128)
void gemm_wmma(const __bf16* __restrict__ A, const __bf16* __restrict__ W,
               void* __restrict__ Cout, int M, int N, int K) {
  __shared__ __attribute__((aligned(32))) __bf16 wtile[2][64 * 32];

  const int tid  = threadIdx.x;
  const int wave = tid >> 5;
  const int lane = tid & 31;
  const int hlf  = lane >> 4;
  const int r    = lane & 15;
  const int n0 = blockIdx.x * 64;
  const int m0 = blockIdx.y * 64 + wave * 16;
  const __bf16* Arow = A + (size_t)(m0 + r) * K;

  // Cooperative staging: 64 rows x 32 k x 2B = 4 KB = 256 x 16B chunks;
  // each of the 128 threads moves 2 chunks.
  const int u0 = tid, u1 = tid + 128;
  const __bf16* Wp0 = W + (size_t)(n0 + (u0 >> 2)) * K + (u0 & 3) * 8;
  const __bf16* Wp1 = W + (size_t)(n0 + (u1 >> 2)) * K + (u1 & 3) * 8;
  const int l0 = (u0 >> 2) * 32 + (u0 & 3) * 8;
  const int l1 = (u1 >> 2) * 32 + (u1 & 3) * 8;

  v8f acc0 = {0.f,0.f,0.f,0.f,0.f,0.f,0.f,0.f};
  v8f acc1 = acc0, acc2 = acc0, acc3 = acc0;

  // Prologue: stage first slab into buffer 0.
  stage16(Wp0, &wtile[0][l0]);
  stage16(Wp1, &wtile[0][l1]);

  int ib = 0;
  for (int kk = 0; kk < K; kk += 32, ib ^= 1) {
    stage_wait();          // own wave's async chunks have landed
    __syncthreads();       // everyone's chunks have landed
    if (kk + 32 < K) {     // prefetch next slab into the other buffer
      stage16(Wp0 + kk + 32, &wtile[ib ^ 1][l0]);
      stage16(Wp1 + kk + 32, &wtile[ib ^ 1][l1]);
    }
    __builtin_prefetch(Arow + kk + 256, 0, 3);       // global_prefetch_b8

    v16bf a = load_afrag(Arow, kk + hlf * 16);
    const __bf16* wt = &wtile[ib][0];
    v16bf b0 = load_bfrag(wt + (      r) * 32, hlf * 16);
    v16bf b1 = load_bfrag(wt + (16 + r) * 32, hlf * 16);
    v16bf b2 = load_bfrag(wt + (32 + r) * 32, hlf * 16);
    v16bf b3 = load_bfrag(wt + (48 + r) * 32, hlf * 16);
    acc0 = wmma_bf16(a, b0, acc0);
    acc1 = wmma_bf16(a, b1, acc1);
    acc2 = wmma_bf16(a, b2, acc2);
    acc3 = wmma_bf16(a, b3, acc3);
    __syncthreads();       // reads of wtile[ib] done before it is re-staged
  }

  v8f accs[4] = {acc0, acc1, acc2, acc3};
  #pragma unroll
  for (int t = 0; t < 4; t++) {
    const int col = n0 + t * 16 + r;
    #pragma unroll
    for (int v = 0; v < 8; v++) {
      const int m = m0 + v + 8 * hlf;        // C layout: row = vgpr + 8*half
      const float val = accs[t][v];
      if (OUT_BF16) ((__bf16*)Cout)[(size_t)m * N + col] = (__bf16)val;
      else          ((float*)Cout)[(size_t)m * N + col]  = val;
    }
  }
}

// ---------------------------------------------------------------------------
// RMSNorm row kernel: out = in * rsqrt(mean(in^2)+eps) * w   (f32 -> bf16)
// ---------------------------------------------------------------------------
__global__ __launch_bounds__(256)
void rmsnorm_bf16(const float* __restrict__ in, const float* __restrict__ w,
                  __bf16* __restrict__ out, int C, int in_stride, int out_stride) {
  __shared__ float red[8];
  __shared__ float sc_s;
  const int row = blockIdx.x;
  const float* ir = in + (size_t)row * in_stride;
  float ss = 0.f;
  for (int c = threadIdx.x; c < C; c += 256) { float v = ir[c]; ss += v * v; }
  #pragma unroll
  for (int o = 16; o >= 1; o >>= 1) ss += __shfl_xor(ss, o, 32);
  if ((threadIdx.x & 31) == 0) red[threadIdx.x >> 5] = ss;
  __syncthreads();
  if (threadIdx.x == 0) {
    float t = 0.f;
    for (int i = 0; i < 8; i++) t += red[i];
    sc_s = rsqrtf(t / (float)C + 1e-6f);
  }
  __syncthreads();
  const float sc = sc_s;
  for (int c = threadIdx.x; c < C; c += 256)
    out[(size_t)row * out_stride + c] = (__bf16)(ir[c] * sc * w[c]);
}

// ---------------------------------------------------------------------------
// Q prep: read q (BS, H*192) bf16, apply interleaved RoPE to last 64 dims,
// write Qb as (b, h, s, 192) bf16.
// ---------------------------------------------------------------------------
__global__ __launch_bounds__(256)
void prep_q(const __bf16* __restrict__ q, const float* __restrict__ sint,
            const float* __restrict__ cost, __bf16* __restrict__ Qb) {
  const int m  = blockIdx.x;
  const int sp = m & (Sc - 1);
  const int b  = m >> 10;
  const __bf16* qr = q + (size_t)m * (Hc * DQKc);
  for (int idx = threadIdx.x; idx < Hc * DQKc; idx += 256) {
    const int h = idx / DQKc, d = idx - h * DQKc;
    float val;
    if (d < NOPEc) {
      val = (float)qr[idx];
    } else {
      const int i = (d - NOPEc) >> 1;
      const float e = (float)qr[h * DQKc + NOPEc + 2 * i];
      const float o = (float)qr[h * DQKc + NOPEc + 2 * i + 1];
      const float s = sint[sp * (ROPEc / 2) + i];
      const float c = cost[sp * (ROPEc / 2) + i];
      val = ((d & 1) == 0) ? (e * c - o * s) : (e * s + o * c);
    }
    Qb[(((size_t)b * Hc + h) * Sc + sp) * DQKc + d] = (__bf16)val;
  }
}

// ---------------------------------------------------------------------------
// K prep: concat k_nope (BS, H*128) bf16 with RoPE'd k_pe from kv f32
// (broadcast across heads); write Kb (b, h, s, 192) bf16.
// ---------------------------------------------------------------------------
__global__ __launch_bounds__(256)
void prep_k(const __bf16* __restrict__ kn, const float* __restrict__ kvf,
            const float* __restrict__ sint, const float* __restrict__ cost,
            __bf16* __restrict__ Kb) {
  const int m  = blockIdx.x;
  const int sp = m & (Sc - 1);
  const int b  = m >> 10;
  for (int idx = threadIdx.x; idx < Hc * DQKc; idx += 256) {
    const int h = idx / DQKc, d = idx - h * DQKc;
    float val;
    if (d < NOPEc) {
      val = (float)kn[(size_t)m * (Hc * NOPEc) + h * NOPEc + d];
    } else {
      const int i = (d - NOPEc) >> 1;
      const float e = kvf[(size_t)m * (KVLRc + ROPEc) + KVLRc + 2 * i];
      const float o = kvf[(size_t)m * (KVLRc + ROPEc) + KVLRc + 2 * i + 1];
      const float s = sint[sp * (ROPEc / 2) + i];
      const float c = cost[sp * (ROPEc / 2) + i];
      val = ((d & 1) == 0) ? (e * c - o * s) : (e * s + o * c);
    }
    Kb[(((size_t)b * Hc + h) * Sc + sp) * DQKc + d] = (__bf16)val;
  }
}

// ---------------------------------------------------------------------------
// V prep: transpose (BS, H*128) bf16 -> Vt (b, h, 128, s) bf16 so the P*V
// B-operand is contiguous over keys.
// ---------------------------------------------------------------------------
__global__ __launch_bounds__(256)
void prep_v(const __bf16* __restrict__ vin, __bf16* __restrict__ Vt) {
  const int m  = blockIdx.x;
  const int sp = m & (Sc - 1);
  const int b  = m >> 10;
  for (int idx = threadIdx.x; idx < Hc * VDc; idx += 256) {
    const int h = idx >> 7, d = idx & 127;
    Vt[(((size_t)b * Hc + h) * VDc + d) * Sc + sp] =
        vin[(size_t)m * (Hc * VDc) + idx];
  }
}

// ---------------------------------------------------------------------------
// Flash-style causal attention.  4 waves / block, each wave owns one
// (b, h, 16-query) tile.  Per 32-key step: 12 score WMMAs (d_qk=192), online
// softmax (cross-lane reductions within 16-lane halves, matching the C-matrix
// row distribution), P through per-wave LDS to A-layout, 8 P*V WMMAs into a
// 16x128 f32 accumulator.  No block barriers -> divergent trip counts OK.
// ---------------------------------------------------------------------------
__global__ __launch_bounds__(128)
void mla_attention(const __bf16* __restrict__ Qb, const __bf16* __restrict__ Kb,
                   const __bf16* __restrict__ Vt, __bf16* __restrict__ Ob) {
  __shared__ __attribute__((aligned(32))) __bf16 pLdsAll[4][16 * 32];

  const int wave = threadIdx.x >> 5;
  const int tile = blockIdx.x * 4 + wave;
  const int qt = tile & 63;
  const int bh = tile >> 6;
  const int b  = bh >> 4;
  const int h  = bh & 15;
  const int q0 = qt * 16;
  const int lane = threadIdx.x & 31;
  const int hlf  = lane >> 4;
  const int r    = lane & 15;
  const float scl = 0.07216878364870323f;    // 1/sqrt(192)
  __bf16* pLds = pLdsAll[wave];

  // Preload the 16x192 Q tile as six A-fragments.
  const __bf16* Qrow = Qb + ((size_t)bh * Sc + q0 + r) * DQKc;
  v16bf qf[6];
  #pragma unroll
  for (int ks = 0; ks < 6; ks++) qf[ks] = load_afrag(Qrow, ks * 32 + hlf * 16);

  const v8f vzero = {0.f,0.f,0.f,0.f,0.f,0.f,0.f,0.f};
  v8f acc[8];
  float mrow[8], lrow[8];
  #pragma unroll
  for (int t = 0; t < 8; t++) acc[t] = vzero;
  #pragma unroll
  for (int v = 0; v < 8; v++) { mrow[v] = -1e30f; lrow[v] = 0.f; }

  for (int k0 = 0; k0 < q0 + 16; k0 += 32) {
    // ---- scores: S = Q * K^T over d_qk=192 (two 16-col tiles) ----
    v8f s0 = vzero, s1 = vzero;
    const __bf16* Kr0 = Kb + ((size_t)bh * Sc + k0 + r) * DQKc;
    const __bf16* Kr1 = Kr0 + (size_t)16 * DQKc;
    #pragma unroll
    for (int ks = 0; ks < 6; ks++) {
      const int kb = ks * 32 + hlf * 16;
      s0 = wmma_bf16(qf[ks], load_bfrag(Kr0, kb), s0);
      s1 = wmma_bf16(qf[ks], load_bfrag(Kr1, kb), s1);
    }

    // ---- causal mask + online softmax update ----
    float corrv[8];
    #pragma unroll
    for (int v = 0; v < 8; v++) {
      const int qr = q0 + v + 8 * hlf;       // this lane's query row for vgpr v
      float a0 = s0[v] * scl; if (k0 + r > qr)      a0 = -1e30f;
      float a1 = s1[v] * scl; if (k0 + 16 + r > qr) a1 = -1e30f;
      float tm = fmaxf(a0, a1);
      #pragma unroll
      for (int msk = 8; msk >= 1; msk >>= 1) tm = fmaxf(tm, __shfl_xor(tm, msk, 32));
      const float mn   = fmaxf(mrow[v], tm);
      const float corr = __expf(mrow[v] - mn);
      const float p0 = __expf(a0 - mn);
      const float p1 = __expf(a1 - mn);
      float rs = p0 + p1;
      #pragma unroll
      for (int msk = 8; msk >= 1; msk >>= 1) rs += __shfl_xor(rs, msk, 32);
      lrow[v] = lrow[v] * corr + rs;
      mrow[v] = mn;
      corrv[v] = corr;
      pLds[(v + 8 * hlf) * 32 + r]      = (__bf16)p0;
      pLds[(v + 8 * hlf) * 32 + 16 + r] = (__bf16)p1;
    }
    #pragma unroll
    for (int t = 0; t < 8; t++)
      #pragma unroll
      for (int v = 0; v < 8; v++)
        acc[t][v] *= corrv[v];

    asm volatile("s_wait_dscnt 0x0" ::: "memory");   // P tile visible in LDS

    // Re-read P in A-operand layout.
    BF16x16 pt;
    pt.h[0] = *(const v8bf*)(pLds + r * 32 + hlf * 16);
    pt.h[1] = *(const v8bf*)(pLds + r * 32 + 8 + hlf * 16);
    const v16bf pf = pt.v;

    // ---- O += P * V (8 column tiles of 16) ----
    #pragma unroll
    for (int t = 0; t < 8; t++) {
      const __bf16* Vr =
          Vt + ((size_t)bh * VDc + t * 16 + r) * Sc + k0 + hlf * 16;
      acc[t] = wmma_bf16(pf, *(const v16bf*)Vr, acc[t]);
    }
  }

  // ---- epilogue: divide by row sums, write (b*s, h*128+d) bf16 ----
  #pragma unroll
  for (int t = 0; t < 8; t++) {
    #pragma unroll
    for (int v = 0; v < 8; v++) {
      const int qr = q0 + v + 8 * hlf;
      const float o = acc[t][v] / lrow[v];
      Ob[((size_t)(b * Sc + qr)) * (Hc * VDc) + h * VDc + t * 16 + r] = (__bf16)o;
    }
  }
}

// ---------------------------------------------------------------------------
// Host launcher
// ---------------------------------------------------------------------------
extern "C" void kernel_launch(void* const* d_in, const int* in_sizes, int n_in,
                              void* d_out, int out_size, void* d_ws, size_t ws_size,
                              hipStream_t stream) {
  (void)in_sizes; (void)n_in; (void)out_size; (void)ws_size;
  const float* x        = (const float*)d_in[0];
  const float* wq_a     = (const float*)d_in[1];
  const float* w_qa_ln  = (const float*)d_in[2];
  const float* wq_b     = (const float*)d_in[3];
  const float* wkv_a    = (const float*)d_in[4];
  const float* w_kva_ln = (const float*)d_in[5];
  const float* wk_b     = (const float*)d_in[6];
  const float* wv_b     = (const float*)d_in[7];
  const float* wo       = (const float*)d_in[8];
  const float* rsin     = (const float*)d_in[9];
  const float* rcos     = (const float*)d_in[10];
  // d_in[11] = seqstarts: layout is uniform (S=1024 per batch), not needed.

  char* base = (char*)d_ws;
  size_t off = 0;
  auto take = [&](size_t bytes) -> void* {
    void* p = base + off;
    off += (bytes + 255) & ~(size_t)255;
    return p;
  };

  __bf16* xb    = (__bf16*)take((size_t)BSc * HIDc * 2);
  __bf16* wqab  = (__bf16*)take((size_t)QLRc * HIDc * 2);
  __bf16* wqbb  = (__bf16*)take((size_t)Hc * DQKc * QLRc * 2);
  __bf16* wkvab = (__bf16*)take((size_t)(KVLRc + ROPEc) * HIDc * 2);
  __bf16* wkbb  = (__bf16*)take((size_t)Hc * NOPEc * KVLRc * 2);
  __bf16* wvbb  = (__bf16*)take((size_t)Hc * VDc * KVLRc * 2);
  __bf16* wob   = (__bf16*)take((size_t)HIDc * Hc * VDc * 2);
  float*  tq    = (float*)take((size_t)BSc * QLRc * 4);   // reused below
  __bf16* qfull = (__bf16*)tq;   // BSc*3072 bf16 == BSc*1536 f32 bytes
  __bf16* qlat  = (__bf16*)take((size_t)BSc * QLRc * 2);
  float*  kvf   = (float*)take((size_t)BSc * (KVLRc + ROPEc) * 4);
  __bf16* ckvb  = (__bf16*)take((size_t)BSc * KVLRc * 2);
  __bf16* knb   = (__bf16*)take((size_t)BSc * Hc * NOPEc * 2);
  __bf16* vb    = (__bf16*)take((size_t)BSc * Hc * VDc * 2);
  __bf16* Qb    = (__bf16*)take((size_t)BSc * Hc * DQKc * 2);
  __bf16* Kb    = (__bf16*)take((size_t)BSc * Hc * DQKc * 2);
  __bf16* Vt    = (__bf16*)take((size_t)BSc * Hc * VDc * 2);
  __bf16* Ob    = (__bf16*)take((size_t)BSc * Hc * VDc * 2);

  auto cg = [](int n) { return dim3((unsigned)((n + 255) / 256)); };

  // f32 -> bf16 conversions
  cvt_bf16<<<cg(BSc * HIDc), 256, 0, stream>>>(x, xb, BSc * HIDc);
  cvt_bf16<<<cg(QLRc * HIDc), 256, 0, stream>>>(wq_a, wqab, QLRc * HIDc);
  cvt_bf16<<<cg(Hc * DQKc * QLRc), 256, 0, stream>>>(wq_b, wqbb, Hc * DQKc * QLRc);
  cvt_bf16<<<cg((KVLRc + ROPEc) * HIDc), 256, 0, stream>>>(wkv_a, wkvab,
                                                           (KVLRc + ROPEc) * HIDc);
  cvt_bf16<<<cg(Hc * NOPEc * KVLRc), 256, 0, stream>>>(wk_b, wkbb, Hc * NOPEc * KVLRc);
  cvt_bf16<<<cg(Hc * VDc * KVLRc), 256, 0, stream>>>(wv_b, wvbb, Hc * VDc * KVLRc);
  cvt_bf16<<<cg(HIDc * Hc * VDc), 256, 0, stream>>>(wo, wob, HIDc * Hc * VDc);

  // Q chain
  gemm_wmma<false><<<dim3(QLRc / 64, BSc / 64), 128, 0, stream>>>(
      xb, wqab, tq, BSc, QLRc, HIDc);
  rmsnorm_bf16<<<BSc, 256, 0, stream>>>(tq, w_qa_ln, qlat, QLRc, QLRc, QLRc);
  gemm_wmma<true><<<dim3(Hc * DQKc / 64, BSc / 64), 128, 0, stream>>>(
      qlat, wqbb, qfull, BSc, Hc * DQKc, QLRc);

  // KV chain
  gemm_wmma<false><<<dim3((KVLRc + ROPEc) / 64, BSc / 64), 128, 0, stream>>>(
      xb, wkvab, kvf, BSc, KVLRc + ROPEc, HIDc);
  rmsnorm_bf16<<<BSc, 256, 0, stream>>>(kvf, w_kva_ln, ckvb, KVLRc,
                                        KVLRc + ROPEc, KVLRc);
  gemm_wmma<true><<<dim3(Hc * NOPEc / 64, BSc / 64), 128, 0, stream>>>(
      ckvb, wkbb, knb, BSc, Hc * NOPEc, KVLRc);
  gemm_wmma<true><<<dim3(Hc * VDc / 64, BSc / 64), 128, 0, stream>>>(
      ckvb, wvbb, vb, BSc, Hc * VDc, KVLRc);

  // RoPE + repack
  prep_q<<<BSc, 256, 0, stream>>>(qfull, rsin, rcos, Qb);
  prep_k<<<BSc, 256, 0, stream>>>(knb, kvf, rsin, rcos, Kb);
  prep_v<<<BSc, 256, 0, stream>>>(vb, Vt);

  // Attention (4 q-tiles per 128-thread block)
  mla_attention<<<Bc * Hc * (Sc / 16) / 4, 128, 0, stream>>>(Qb, Kb, Vt, Ob);

  // Output projection -> f32 d_out
  gemm_wmma<false><<<dim3(HIDc / 64, BSc / 64), 128, 0, stream>>>(
      Ob, wob, (float*)d_out, BSc, HIDc, Hc * VDc);
}
